// Network_70300024701119
// MI455X (gfx1250) — compile-verified, matrix operand order checked by reading, not run on hardware
//
#include <hip/hip_runtime.h>
#include <hip/hip_bf16.h>
#include <cstdint>
#include <cstddef>

// ---------------------------------------------------------------------------
// PointNet++ SSG + PointNet + critic forward for MI455X (gfx1250, wave32).
// All dense layers go through one wave-level WMMA GEMM kernel
// (v_wmma_f32_16x16x32_f16, f32 accumulate). Serial selection stages (FPS,
// top-50) run in LDS with block reductions. The last FP module + fc are
// evaluated only at point 0 per batch, since only whole[:,0,:] is consumed.
//
// Round-1 fix: GEMM fragment loads are now unconditional (row-clamped; K-tail
// handled with index clamp + select-zero) so the compiler batches the 32
// global_load_b32s per K-step behind a single s_wait_loadcnt instead of
// emitting per-element exec-masked loads each followed by a full wait.
// ---------------------------------------------------------------------------

#define BATCH 8
#define NPTS 4096

typedef __attribute__((ext_vector_type(16))) _Float16 v16h;
typedef __attribute__((ext_vector_type(8)))  float    v8f;

// ---------------------------- WMMA GEMM ------------------------------------
// Y[M,N] = act( bn( X[M,K] * W[N,K]^T + bias ) )
// act: 0 = none, 1 = relu, 2 = leaky(0.01)
// bn (if g != null): y = y * (g / sqrt(1+1e-5)) + b

__device__ __forceinline__ int frag_k(int idx, int hi) {
  // 16-bit A-matrix 16x32 lane layout (ISA 7.12.2):
  // idx 0..7  -> K = 8*hi + idx
  // idx 8..15 -> K = 16 + 8*hi + (idx-8)
  return ((idx & 8) << 1) + (hi << 3) + (idx & 7);
}

__global__ void wmma_gemm_kernel(const float* __restrict__ X,
                                 const float* __restrict__ W,
                                 const float* __restrict__ bias,
                                 const float* __restrict__ bng,
                                 const float* __restrict__ bnb,
                                 float* __restrict__ Y,
                                 int M, int N, int K, int act) {
  const int tilesN = (N + 15) >> 4;
  const int tilesM = (M + 15) >> 4;
  int wave = (blockIdx.x * blockDim.x + threadIdx.x) >> 5;
  if (wave >= tilesM * tilesN) return;   // whole-wave uniform exit (EXEC stays all-1 for WMMA)
  int tm = wave / tilesN;
  int tn = wave % tilesN;
  int lane = threadIdx.x & 31;
  int r  = lane & 15;
  int hi = lane >> 4;

  // Row-clamped base pointers: out-of-range rows read row (M-1)/(N-1); their
  // products only reach output rows/cols that the epilogue never stores.
  int rowA = tm * 16 + r;
  int rowB = tn * 16 + r;
  int rowAc = rowA < M ? rowA : (M - 1);
  int rowBc = rowB < N ? rowB : (N - 1);
  const float* Xr = X + (size_t)rowAc * K;
  const float* Wr = W + (size_t)rowBc * K;

  v8f acc = {};
  const int kfull = K & ~31;
  int k0 = 0;
  for (; k0 < kfull; k0 += 32) {
    if (k0 + 32 < kfull) {  // prefetch next K-step (global_prefetch_b8)
      __builtin_prefetch(Xr + k0 + 32, 0, 0);
      __builtin_prefetch(Wr + k0 + 32, 0, 0);
    }
    v16h a, b;
#pragma unroll
    for (int i = 0; i < 16; ++i) {
      int kk = k0 + frag_k(i, hi);
      a[i] = (_Float16)Xr[kk];   // unconditional, batched loads
      b[i] = (_Float16)Wr[kk];
    }
    acc = __builtin_amdgcn_wmma_f32_16x16x32_f16(
        /*neg_a=*/false, a, /*neg_b=*/false, b,
        /*c_mod=*/(short)0, acc, /*reuse_a=*/false, /*reuse_b=*/false);
  }
  if (k0 < K) {
    // K tail: clamp index so the load stays unconditional, then select-zero
    // both operands at padded k (keeps the accumulation exact, no exec masks).
    v16h a, b;
#pragma unroll
    for (int i = 0; i < 16; ++i) {
      int kk = k0 + frag_k(i, hi);
      int kc = kk < K ? kk : 0;
      float xa = Xr[kc];
      float xb = Wr[kc];
      a[i] = kk < K ? (_Float16)xa : (_Float16)0.f;
      b[i] = kk < K ? (_Float16)xb : (_Float16)0.f;
    }
    acc = __builtin_amdgcn_wmma_f32_16x16x32_f16(
        false, a, false, b, (short)0, acc, false, false);
  }

  // C/D layout: lane&15 = column; row = vgpr + 8*(lane>>4)
  const float invs = 0.999995000037499684f;  // 1/sqrt(1 + 1e-5)
  int n = tn * 16 + r;
  if (n < N) {
    float addb = bias ? bias[n] : 0.f;
    float gs = bng ? bng[n] * invs : 1.f;
    float bb = bnb ? bnb[n] : 0.f;
#pragma unroll
    for (int v = 0; v < 8; ++v) {
      int m = tm * 16 + v + (hi << 3);
      if (m < M) {
        float val = acc[v] + addb;
        if (bng) val = val * gs + bb;
        if (act == 1) val = val > 0.f ? val : 0.f;
        else if (act == 2) val = val > 0.f ? val : 0.01f * val;
        Y[(size_t)m * N + n] = val;
      }
    }
  }
}

// ---------------------------- FPS ------------------------------------------
// One block per batch. Coordinates + running min-distance kept on chip:
// coords in LDS, per-point distances in registers (strided), block argmax
// with first-index tie-break (matches jnp.argmax).

__global__ void k_fps(const float* __restrict__ xyz, int N, int S,
                      int* __restrict__ outIdx, float* __restrict__ newXyz) {
  __shared__ float sx[NPTS], sy[NPTS], sz[NPTS];
  __shared__ float rbest[256];
  __shared__ int ribest[256];
  __shared__ int sfar;
  int b = blockIdx.x, t = threadIdx.x;
  const float* base = xyz + (size_t)b * N * 3;
  for (int p = t; p < N; p += 256) {
    sx[p] = base[p * 3 + 0];
    sy[p] = base[p * 3 + 1];
    sz[p] = base[p * 3 + 2];
  }
  float dist[16];
#pragma unroll
  for (int c = 0; c < 16; ++c) dist[c] = 1e10f;
  if (t == 0) sfar = 0;
  __syncthreads();

  int chunks = (N + 255) / 256;
  for (int i = 0; i < S; ++i) {
    int far = sfar;
    if (t == 0) {
      outIdx[b * S + i] = far;
      newXyz[((size_t)b * S + i) * 3 + 0] = sx[far];
      newXyz[((size_t)b * S + i) * 3 + 1] = sy[far];
      newXyz[((size_t)b * S + i) * 3 + 2] = sz[far];
    }
    float fx = sx[far], fy = sy[far], fz = sz[far];
    float best = -1.f;
    int bidx = 0;
    for (int c = 0; c < chunks; ++c) {
      int p = c * 256 + t;
      if (p < N) {
        float dx = sx[p] - fx, dy = sy[p] - fy, dz = sz[p] - fz;
        float d = dx * dx + dy * dy + dz * dz;
        float nd = fminf(dist[c], d);
        dist[c] = nd;
        if (nd > best) { best = nd; bidx = p; }   // strict > keeps lowest index
      }
    }
    rbest[t] = best; ribest[t] = bidx;
    __syncthreads();
    for (int s2 = 128; s2 > 0; s2 >>= 1) {
      if (t < s2) {
        float o = rbest[t + s2]; int oi = ribest[t + s2];
        if (o > rbest[t] || (o == rbest[t] && oi < ribest[t])) {
          rbest[t] = o; ribest[t] = oi;
        }
      }
      __syncthreads();
    }
    if (t == 0) sfar = ribest[0];
    __syncthreads();
  }
}

// ---------------------------- Ball query -----------------------------------
// Reference semantics: nsample lowest indices among points with d2 < r2,
// remainder filled with the first hit (or all zeros if none).

__global__ void k_ballquery(const float* __restrict__ xyz,
                            const float* __restrict__ newxyz,
                            int N, int S, float r2, int ns,
                            int* __restrict__ idx, int total) {
  int gid = blockIdx.x * 256 + threadIdx.x;
  if (gid >= total) return;
  int b = gid / S;
  const float* q = newxyz + (size_t)gid * 3;
  const float* base = xyz + (size_t)b * N * 3;
  int* o = idx + (size_t)gid * ns;
  int cnt = 0;
  for (int j = 0; j < N && cnt < ns; ++j) {
    float dx = base[j * 3 + 0] - q[0];
    float dy = base[j * 3 + 1] - q[1];
    float dz = base[j * 3 + 2] - q[2];
    if (dx * dx + dy * dy + dz * dz < r2) o[cnt++] = j;
  }
  int first = (cnt > 0) ? o[0] : 0;
  for (int j = cnt; j < ns; ++j) o[j] = first;
}

// ---------------------------- Group + concat -------------------------------
__global__ void k_group(const float* __restrict__ xyz,
                        const float* __restrict__ feats, int C,
                        const float* __restrict__ newxyz,
                        const int* __restrict__ idx,
                        int N, int S, int ns, float* __restrict__ h, int total) {
  int gid = blockIdx.x * 256 + threadIdx.x;
  if (gid >= total) return;               // total = B*S*ns rows
  int bs = gid / ns;
  int b = bs / S;
  int j = idx[gid];
  float* o = h + (size_t)gid * (3 + C);
  const float* q = newxyz + (size_t)bs * 3;
  const float* pt = xyz + ((size_t)b * N + j) * 3;
  o[0] = pt[0] - q[0];
  o[1] = pt[1] - q[1];
  o[2] = pt[2] - q[2];
  const float* f = feats + ((size_t)b * N + j) * C;
  for (int c = 0; c < C; ++c) o[3 + c] = f[c];
}

// ---------------------------- Max over a group of rows ---------------------
__global__ void k_max_over(const float* __restrict__ in, int gsize, int C,
                           float* __restrict__ out, int total) {
  int gid = blockIdx.x * 256 + threadIdx.x;
  if (gid >= total) return;               // total = groups*C
  int c = gid % C;
  int g = gid / C;
  const float* p = in + (size_t)g * gsize * C + c;
  float m = p[0];
  for (int j = 1; j < gsize; ++j) m = fmaxf(m, p[(size_t)j * C]);
  out[gid] = m;
}

// ---------------------------- 3-NN for FP interpolation --------------------
__global__ void k_knn3(const float* __restrict__ ux, const float* __restrict__ kx,
                       int U, int Kp, int* __restrict__ idx3,
                       float* __restrict__ w3, int total) {
  int gid = blockIdx.x * 256 + threadIdx.x;
  if (gid >= total) return;               // total = B*U
  int b = gid / U;
  const float* q = ux + (size_t)gid * 3;
  const float* base = kx + (size_t)b * Kp * 3;
  float d0 = 1e30f, d1 = 1e30f, d2 = 1e30f;
  int i0 = 0, i1 = 0, i2 = 0;
  for (int j = 0; j < Kp; ++j) {
    float dx = q[0] - base[j * 3 + 0];
    float dy = q[1] - base[j * 3 + 1];
    float dz = q[2] - base[j * 3 + 2];
    float d = dx * dx + dy * dy + dz * dz;
    if (d < d0)      { d2 = d1; i2 = i1; d1 = d0; i1 = i0; d0 = d; i0 = j; }
    else if (d < d1) { d2 = d1; i2 = i1; d1 = d;  i1 = j; }
    else if (d < d2) { d2 = d;  i2 = j; }
  }
  float w0 = 1.f / (d0 + 1e-8f);
  float w1 = 1.f / (d1 + 1e-8f);
  float w2 = 1.f / (d2 + 1e-8f);
  float inv = 1.f / (w0 + w1 + w2);
  idx3[gid * 3 + 0] = i0; idx3[gid * 3 + 1] = i1; idx3[gid * 3 + 2] = i2;
  w3[gid * 3 + 0] = w0 * inv; w3[gid * 3 + 1] = w1 * inv; w3[gid * 3 + 2] = w2 * inv;
}

// ---------------------------- Interp + concat ------------------------------
__global__ void k_interp_concat(const float* __restrict__ kfeat, int Kp, int Ck,
                                const int* __restrict__ idx3,
                                const float* __restrict__ w3,
                                const float* __restrict__ ufeat, int Cu,
                                int U, float* __restrict__ h, long total) {
  long gid = (long)blockIdx.x * 256 + threadIdx.x;
  if (gid >= total) return;               // total = B*U*(Ck+Cu)
  int Ct = Ck + Cu;
  long row = gid / Ct;
  int c = (int)(gid % Ct);
  int b = (int)(row / U);
  float v;
  if (c < Ck) {
    float acc = 0.f;
    for (int j = 0; j < 3; ++j)
      acc += w3[row * 3 + j] * kfeat[((size_t)b * Kp + idx3[row * 3 + j]) * Ck + c];
    v = acc;
  } else {
    v = ufeat[(size_t)row * Cu + (c - Ck)];
  }
  h[(size_t)row * Ct + c] = v;
}

// ---------------------------- disf / fdis ----------------------------------
__global__ void k_disf(const float* __restrict__ scene, const float* __restrict__ robot,
                       const unsigned char* __restrict__ gt,
                       float* __restrict__ disf, float* __restrict__ fdis,
                       int N, int total) {
  int gid = blockIdx.x * 256 + threadIdx.x;
  if (gid >= total) return;
  int b = gid / N;
  const float* p  = scene + (size_t)gid * 3;
  const float* p0 = scene + (size_t)b * N * 3;
  float rx = robot[b * 3 + 0], ry = robot[b * 3 + 1], rz = robot[b * 3 + 2] + 1.f;
  float v1x = p[0] - p0[0], v1y = p[1] - p0[1], v1z = p[2] - p0[2];
  float v2x = p[0] - rx,    v2y = p[1] - ry,    v2z = p[2] - rz;
  float cx = v1y * v2z - v1z * v2y;
  float cy = v1z * v2x - v1x * v2z;
  float cz = v1x * v2y - v1y * v2x;
  disf[gid * 3 + 0] = cx;
  disf[gid * 3 + 1] = cy;
  disf[gid * 3 + 2] = cz;
  float f = sqrtf(cx * cx + cy * cy + cz * cz);
  fdis[gid] = gt[gid] ? 1e4f : f;
}

// ---------------------------- top-50 smallest + gather ---------------------
__global__ void k_top50(const float* __restrict__ fdis, const float* __restrict__ disf,
                        float* __restrict__ disf_pcs, int N) {
  __shared__ float sd[NPTS];
  __shared__ float rb[256];
  __shared__ int ri[256];
  __shared__ int sel[50];
  int b = blockIdx.x, t = threadIdx.x;
  for (int p = t; p < N; p += 256) sd[p] = fdis[(size_t)b * N + p];
  __syncthreads();
  for (int i = 0; i < 50; ++i) {
    float best = 1e30f; int bi = 0;
    for (int p = t; p < N; p += 256) {
      float v = sd[p];
      if (v < best) { best = v; bi = p; }   // strict < keeps lowest index
    }
    rb[t] = best; ri[t] = bi;
    __syncthreads();
    for (int s2 = 128; s2 > 0; s2 >>= 1) {
      if (t < s2) {
        float o = rb[t + s2]; int oi = ri[t + s2];
        if (o < rb[t] || (o == rb[t] && oi < ri[t])) { rb[t] = o; ri[t] = oi; }
      }
      __syncthreads();
    }
    if (t == 0) { sel[i] = ri[0]; sd[ri[0]] = 1e30f; }
    __syncthreads();
  }
  for (int e = t; e < 50 * 3; e += 256) {
    int j = e / 3, d = e % 3;
    disf_pcs[((size_t)b * 50 + j) * 3 + d] = disf[((size_t)b * N + sel[j]) * 3 + d];
  }
}

// ---------------------------- STN helpers ----------------------------------
__global__ void k_add_eye(float* __restrict__ m, int k, int total) {
  int gid = blockIdx.x * 256 + threadIdx.x;
  if (gid >= total) return;               // total = B*k
  int i = gid % k, b = gid / k;
  m[(size_t)b * k * k + (size_t)i * k + i] += 1.f;
}

__global__ void k_apply_trans(const float* __restrict__ x, const float* __restrict__ tr,
                              int Np, int C, float* __restrict__ y, int total) {
  int gid = blockIdx.x * 256 + threadIdx.x;
  if (gid >= total) return;               // total = B*Np*C
  int d = gid % C;
  int row = gid / C;
  int b = row / Np;
  const float* xr = x + (size_t)row * C;
  const float* tb = tr + (size_t)b * C * C;
  float acc = 0.f;
  for (int c = 0; c < C; ++c) acc += xr[c] * tb[(size_t)c * C + d];
  y[gid] = acc;
}

// ---------------------------- small glue kernels ---------------------------
__global__ void k_copy_row0(const float* __restrict__ scene, int N,
                            float* __restrict__ out, int total) {
  int gid = blockIdx.x * 256 + threadIdx.x;
  if (gid >= total) return;               // total = B*3
  out[gid] = scene[(size_t)(gid / 3) * N * 3 + (gid % 3)];
}

__global__ void k_concat_all(const float* __restrict__ whole0,
                             const float* __restrict__ gfeat,
                             const float* __restrict__ robot,
                             float* __restrict__ out, int total) {
  int gid = blockIdx.x * 256 + threadIdx.x;
  if (gid >= total) return;               // total = B*259
  int b = gid / 259, c = gid % 259;
  float v;
  if (c < 128)       v = whole0[b * 128 + c];
  else if (c < 256)  v = gfeat[b * 128 + (c - 128)];
  else               v = robot[b * 3 + (c - 256)] + ((c - 256) == 2 ? 1.f : 0.f);
  out[gid] = v;
}

__global__ void k_write_out(const float* __restrict__ logits,
                            const float* __restrict__ gfeat,
                            float* __restrict__ out) {
  int gid = blockIdx.x * 256 + threadIdx.x;
  if (gid < BATCH)                 out[gid] = logits[gid];
  else if (gid < BATCH + BATCH * 128) out[gid] = gfeat[gid - BATCH];
}

// ===========================================================================
// Host side
// ===========================================================================

namespace {

struct LBR {
  const float* W;
  const float* bias;   // linear bias (linb / linbn), null for convbn
  const float* g;      // bn scale, null -> no bn
  const float* b;      // bn shift
  int Cout, Cin;
};

struct Bump {
  char* base;
  size_t off;
  void* get(size_t bytes) {
    size_t a = (off + 255) & ~(size_t)255;
    off = a + bytes;
    return base + a;
  }
  float* f(size_t n) { return (float*)get(n * sizeof(float)); }
  int* i(size_t n) { return (int*)get(n * sizeof(int)); }
};

inline int nblk(long total) { return (int)((total + 255) / 256); }

void gemm(hipStream_t s, const float* X, float* Y, int M, const LBR& L, int act) {
  long tiles = (long)((M + 15) / 16) * ((L.Cout + 15) / 16);
  int blocks = (int)((tiles + 7) / 8);   // 8 waves / 256-thread block
  wmma_gemm_kernel<<<blocks, 256, 0, s>>>(X, L.W, L.bias, L.g, L.b, Y,
                                          M, L.Cout, L.Cin, act);
}

const float* run_chain(hipStream_t s, const float* X, int M, const LBR* Ls, int nL,
                       float* bufA, float* bufB, int act) {
  const float* in = X;
  for (int i = 0; i < nL; ++i) {
    float* out = (in == bufA) ? bufB : bufA;
    gemm(s, in, out, M, Ls[i], act);
    in = out;
  }
  return in;
}

}  // namespace

extern "C" void kernel_launch(void* const* d_in, const int* in_sizes, int n_in,
                              void* d_out, int out_size, void* d_ws, size_t ws_size,
                              hipStream_t stream) {
  (void)in_sizes; (void)n_in; (void)out_size; (void)ws_size;

  const float* scene = (const float*)d_in[0];          // (8,4096,3)
  const float* robot = (const float*)d_in[1];          // (8,3)
  const unsigned char* gt = (const unsigned char*)d_in[2];  // (8,4096) bool

  // ---- unpack params: jax pytree order (sorted dict keys): critic, pn, ssg
  int cur = 3;
  auto F = [&]() { return (const float*)d_in[cur++]; };
  auto linbn = [&](int co, int ci) {  // keys sorted: W, b, bias, g
    LBR L; L.W = F(); L.b = F(); L.bias = F(); L.g = F(); L.Cout = co; L.Cin = ci; return L;
  };
  auto linb = [&](int co, int ci) {   // keys sorted: W, bias
    LBR L; L.W = F(); L.bias = F(); L.g = nullptr; L.b = nullptr; L.Cout = co; L.Cin = ci; return L;
  };
  auto convbn = [&](int co, int ci) { // keys sorted: W, b, g
    LBR L; L.W = F(); L.b = F(); L.g = F(); L.bias = nullptr; L.Cout = co; L.Cin = ci; return L;
  };

  // critic
  LBR c_mlp1 = linb(128, 259);
  LBR c_mlp2 = linb(1, 128);
  // pn (sorted: conv1, conv2, conv3, fc1, fc2, fc3, stn3, stn64)
  LBR conv1 = linbn(64, 3);
  LBR conv2 = linbn(128, 64);
  LBR conv3 = linbn(1024, 128);
  LBR fc1 = linbn(512, 1024);
  LBR fc2 = linbn(256, 512);
  LBR fc3 = linb(128, 256);
  LBR s3_convs[3] = { linbn(64, 3), linbn(128, 64), linbn(1024, 128) };
  LBR s3_fcs[2]   = { linbn(512, 1024), linbn(256, 512) };
  LBR s3_out      = linb(9, 256);
  LBR s64_convs[3] = { linbn(64, 64), linbn(128, 64), linbn(1024, 128) };
  LBR s64_fcs[2]   = { linbn(512, 1024), linbn(256, 512) };
  LBR s64_out      = linb(4096, 256);
  // ssg (sorted: fc, fp, sa)
  LBR ssg_fc = convbn(128, 128);
  LBR fp0[2] = { convbn(256, 768), convbn(256, 256) };
  LBR fp1[2] = { convbn(256, 384), convbn(256, 256) };
  LBR fp2[2] = { convbn(256, 320), convbn(128, 256) };
  LBR fp3[3] = { convbn(128, 131), convbn(128, 128), convbn(128, 128) };
  LBR sa0[3] = { convbn(32, 6),    convbn(32, 32),   convbn(64, 32)  };
  LBR sa1[3] = { convbn(64, 67),   convbn(64, 64),   convbn(128, 64) };
  LBR sa2[3] = { convbn(128, 131), convbn(128, 128), convbn(256, 128) };
  LBR sa3[3] = { convbn(256, 259), convbn(256, 256), convbn(512, 256) };
  const LBR* saL[4] = { sa0, sa1, sa2, sa3 };

  // ---- workspace layout
  Bump ws{ (char*)d_ws, 0 };
  float* bufA = ws.f((size_t)262144 * 64);   // ping
  float* bufB = ws.f((size_t)262144 * 64);   // pong
  float* xyzs[4] = { ws.f(8 * 1024 * 3), ws.f(8 * 256 * 3), ws.f(8 * 64 * 3), ws.f(8 * 16 * 3) };
  float* feats[4] = { ws.f((size_t)8 * 1024 * 64), ws.f((size_t)8 * 256 * 128),
                      ws.f((size_t)8 * 64 * 256),  ws.f((size_t)8 * 16 * 512) };
  float* feats3p = ws.f((size_t)8 * 64 * 256);
  float* feats2p = ws.f((size_t)8 * 256 * 256);
  float* feats1p = ws.f((size_t)8 * 1024 * 128);
  int* fpsIdx = ws.i(8 * 1024);
  int* ballIdx = ws.i((size_t)8 * 1024 * 32);
  int* idx3 = ws.i((size_t)8 * 1024 * 3);
  float* w3 = ws.f((size_t)8 * 1024 * 3);
  float* disf = ws.f((size_t)8 * NPTS * 3);
  float* fdis = ws.f((size_t)8 * NPTS);
  float* disf_pcs = ws.f(8 * 50 * 3);
  float* row0 = ws.f(8 * 3);
  float* pool1024 = ws.f(8 * 1024);
  float* trans9 = ws.f(8 * 9);
  float* transfeat = ws.f((size_t)8 * 4096);
  float* h64 = ws.f((size_t)8 * 50 * 64);
  float* whole0 = ws.f(8 * 128);
  float* gfeat = ws.f(8 * 128);
  float* allfeats = ws.f(8 * 259);
  float* hidden = ws.f(8 * 128);
  float* logits = ws.f(16);

  // =============== PointNet++ SSG: SA modules ===============
  struct SAcfg { int np; float r; int ns; } cfg[4] = {
    {1024, 0.1f, 32}, {256, 0.2f, 32}, {64, 0.4f, 32}, {16, 0.8f, 32} };

  const float* cur_xyz = scene;
  const float* cur_feat = scene;
  int curN = NPTS, curC = 3;
  for (int i = 0; i < 4; ++i) {
    int S = cfg[i].np, ns = cfg[i].ns;
    float r2 = cfg[i].r * cfg[i].r;
    k_fps<<<BATCH, 256, 0, stream>>>(cur_xyz, curN, S, fpsIdx, xyzs[i]);
    int totC = BATCH * S;
    k_ballquery<<<nblk(totC), 256, 0, stream>>>(cur_xyz, xyzs[i], curN, S, r2, ns,
                                                ballIdx, totC);
    long rows = (long)BATCH * S * ns;
    k_group<<<nblk(rows), 256, 0, stream>>>(cur_xyz, cur_feat, curC, xyzs[i],
                                            ballIdx, curN, S, ns, bufA, (int)rows);
    const float* last = run_chain(stream, bufA, (int)rows, saL[i], 3, bufA, bufB, 1);
    int Cout = saL[i][2].Cout;
    long totM = (long)BATCH * S * Cout;
    k_max_over<<<nblk(totM), 256, 0, stream>>>(last, ns, Cout, feats[i], (int)totM);
    cur_xyz = xyzs[i]; cur_feat = feats[i]; curN = S; curC = Cout;
  }

  // =============== FP modules ===============
  // fp0: ux=xyz3(64), kx=xyz4(16), ufeat=feats3(256), kfeat=feats4(512)
  k_knn3<<<nblk(8 * 64), 256, 0, stream>>>(xyzs[2], xyzs[3], 64, 16, idx3, w3, 8 * 64);
  k_interp_concat<<<nblk((long)8 * 64 * 768), 256, 0, stream>>>(
      feats[3], 16, 512, idx3, w3, feats[2], 256, 64, bufA, (long)8 * 64 * 768);
  {
    const float* r = run_chain(stream, bufA, 8 * 64, fp0, 2, bufA, bufB, 1);
    hipMemcpyAsync(feats3p, r, (size_t)8 * 64 * 256 * sizeof(float),
                   hipMemcpyDeviceToDevice, stream);
  }
  // fp1: ux=xyz2(256), kx=xyz3(64), ufeat=feats2(128), kfeat=feats3p(256)
  k_knn3<<<nblk(8 * 256), 256, 0, stream>>>(xyzs[1], xyzs[2], 256, 64, idx3, w3, 8 * 256);
  k_interp_concat<<<nblk((long)8 * 256 * 384), 256, 0, stream>>>(
      feats3p, 64, 256, idx3, w3, feats[1], 128, 256, bufA, (long)8 * 256 * 384);
  {
    const float* r = run_chain(stream, bufA, 8 * 256, fp1, 2, bufA, bufB, 1);
    hipMemcpyAsync(feats2p, r, (size_t)8 * 256 * 256 * sizeof(float),
                   hipMemcpyDeviceToDevice, stream);
  }
  // fp2: ux=xyz1(1024), kx=xyz2(256), ufeat=feats1(64), kfeat=feats2p(256)
  k_knn3<<<nblk(8 * 1024), 256, 0, stream>>>(xyzs[0], xyzs[1], 1024, 256, idx3, w3, 8 * 1024);
  k_interp_concat<<<nblk((long)8 * 1024 * 320), 256, 0, stream>>>(
      feats2p, 256, 256, idx3, w3, feats[0], 64, 1024, bufA, (long)8 * 1024 * 320);
  {
    const float* r = run_chain(stream, bufA, 8 * 1024, fp2, 2, bufA, bufB, 1);
    hipMemcpyAsync(feats1p, r, (size_t)8 * 1024 * 128 * sizeof(float),
                   hipMemcpyDeviceToDevice, stream);
  }
  // fp3 + fc: only point 0 per batch is consumed downstream (whole[:,0,:]).
  k_copy_row0<<<1, 256, 0, stream>>>(scene, NPTS, row0, 8 * 3);
  k_knn3<<<1, 256, 0, stream>>>(row0, xyzs[0], 1, 1024, idx3, w3, 8);
  k_interp_concat<<<nblk(8 * 131), 256, 0, stream>>>(
      feats1p, 1024, 128, idx3, w3, row0, 3, 1, bufA, (long)8 * 131);
  {
    const float* r = run_chain(stream, bufA, 8, fp3, 3, bufA, bufB, 1);
    gemm(stream, r, whole0, 8, ssg_fc, 1);
  }

  // =============== PointNet branch ===============
  k_disf<<<nblk((long)BATCH * NPTS), 256, 0, stream>>>(scene, robot, gt, disf, fdis,
                                                       NPTS, BATCH * NPTS);
  k_top50<<<BATCH, 256, 0, stream>>>(fdis, disf, disf_pcs, NPTS);

  // STN-3
  {
    const float* r = run_chain(stream, disf_pcs, 8 * 50, s3_convs, 3, bufA, bufB, 1);
    k_max_over<<<nblk(8 * 1024), 256, 0, stream>>>(r, 50, 1024, pool1024, 8 * 1024);
    const float* f = run_chain(stream, pool1024, 8, s3_fcs, 2, bufA, bufB, 1);
    gemm(stream, f, trans9, 8, s3_out, 0);
    k_add_eye<<<1, 256, 0, stream>>>(trans9, 3, 8 * 3);
  }
  k_apply_trans<<<nblk(8 * 50 * 3), 256, 0, stream>>>(disf_pcs, trans9, 50, 3, bufA,
                                                      8 * 50 * 3);
  gemm(stream, bufA, h64, 8 * 50, conv1, 1);   // (400,64), kept live across STN-64

  // STN-64
  {
    const float* r = run_chain(stream, h64, 8 * 50, s64_convs, 3, bufA, bufB, 1);
    k_max_over<<<nblk(8 * 1024), 256, 0, stream>>>(r, 50, 1024, pool1024, 8 * 1024);
    const float* f = run_chain(stream, pool1024, 8, s64_fcs, 2, bufA, bufB, 1);
    gemm(stream, f, transfeat, 8, s64_out, 0);
    k_add_eye<<<1, 256, 0, stream>>>(transfeat, 64, 8 * 64);
  }
  k_apply_trans<<<nblk(8 * 50 * 64), 256, 0, stream>>>(h64, transfeat, 50, 64, bufA,
                                                       8 * 50 * 64);
  gemm(stream, bufA, bufB, 8 * 50, conv2, 1);
  gemm(stream, bufB, bufA, 8 * 50, conv3, 0);  // bn, no activation
  k_max_over<<<nblk(8 * 1024), 256, 0, stream>>>(bufA, 50, 1024, pool1024, 8 * 1024);
  gemm(stream, pool1024, bufA, 8, fc1, 1);
  gemm(stream, bufA, bufB, 8, fc2, 1);
  gemm(stream, bufB, gfeat, 8, fc3, 0);

  // =============== Critic ===============
  k_concat_all<<<nblk(8 * 259), 256, 0, stream>>>(whole0, gfeat, robot, allfeats, 8 * 259);
  gemm(stream, allfeats, hidden, 8, c_mlp1, 2);   // leaky_relu(0.01)
  gemm(stream, hidden, logits, 8, c_mlp2, 0);

  k_write_out<<<nblk(8 + 8 * 128), 256, 0, stream>>>(logits, gfeat, (float*)d_out);
}
